// GPT_50483045598097
// MI455X (gfx1250) — compile-verified, matrix operand order checked by reading, not run on hardware
//
#include <hip/hip_runtime.h>
#include <math.h>

typedef __attribute__((ext_vector_type(16))) _Float16 v16h;
typedef __attribute__((ext_vector_type(8)))  _Float16 v8h;
typedef __attribute__((ext_vector_type(8)))  float    v8f;

#define WGSZ 256

__device__ __forceinline__ int   imin(int a, int b) { return a < b ? a : b; }
__device__ __forceinline__ float actf(float v, int act) {
  if (act == 1) return v > 0.f ? v : 0.f;                              // relu
  if (act == 2) return 0.5f * v * (1.f + erff(v * 0.70710678118655f)); // gelu (exact)
  if (act == 3) return 1.f / (1.f + expf(-v));                         // sigmoid
  return v;
}

// ---------------------------------------------------------------------------
// Batched GEMM: C[bz] = act(alpha * A[bz] @ B[bz](^T) + bias)
// A: M x K (row-major, lda), B: K x N (ldb) or N x K if transB, C: M x N (ldc)
// batch bz -> (i = bz / subBatch, j = bz % subBatch); offsets i*s?1 + j*s?2
// 256 threads = 8 wave32; 128x64 C tile; wave w owns rows [w*16, w*16+16) x 64
// cols => 4 WMMA per wave per K-step sharing one A fragment.
// Software pipeline: next K-tile is prefetched from global into registers
// while the current tile is consumed from LDS (global latency overlaps MMA).
// Guarded paths are mask-multiply (no saveexec) behind uniform branches.
// LDS: A K-contiguous, B transposed -> all fragments are aligned ds_load_b128.
// ---------------------------------------------------------------------------
__global__ void __launch_bounds__(WGSZ)
gemm_f16wmma(const float* __restrict__ A, const float* __restrict__ B,
             const float* __restrict__ bias, float* __restrict__ C,
             int M, int N, int K, int lda, int ldb, int ldc,
             long sA1, long sA2, long sB1, long sB2, long sC1, long sC2,
             int subBatch, int transB, int act, float alpha)
{
  {
    int bz = blockIdx.z;
    int bi = bz / subBatch, bj = bz % subBatch;
    A += bi * sA1 + bj * sA2;
    B += bi * sB1 + bj * sB2;
    C += bi * sC1 + bj * sC2;
  }

  // 40-half pitch = 80 B: multiple of 16 B (aligned B128), 20-dword bank stride.
  __shared__ __align__(32) _Float16 As [128][40]; // As[m][k]
  __shared__ __align__(32) _Float16 BsT[64][40];  // BsT[n][k] (transposed tile)

  const int m0   = blockIdx.y * 128;
  const int n0   = blockIdx.x * 64;
  const int tid  = threadIdx.x;
  const int lane = tid & 31;
  const int wave = tid >> 5;          // 0..7 -> M sub-tile of 16 rows

  const bool mFull = (m0 + 128 <= M); // block-uniform
  const bool nFull = (n0 + 64 <= N);  // block-uniform

  // staging coordinates
  const int aRow = tid >> 1;          // 0..127
  const int aKK  = (tid & 1) * 16;    // 0 or 16
  const int bCol = tid >> 2;          // transB: n 0..63
  const int bKK  = (tid & 3) * 8;     // transB: k 0..24
  const int bKrow= tid >> 3;          // !transB: k 0..31
  const int bNN  = (tid & 7) * 8;     // !transB: n 0..56

  // ---- register tile loaders (uniform fast/slow selection) ----
  auto loadA = [&](int k0, v8h& x0, v8h& x1) {
    int gm = m0 + aRow;
    if (mFull && k0 + 32 <= K) {
      const float* Ap = A + (long)gm * lda + (k0 + aKK);
      #pragma unroll
      for (int u = 0; u < 8; ++u) {
        x0[u] = (_Float16)Ap[u];
        x1[u] = (_Float16)Ap[u + 8];
      }
    } else {
      bool rowOk = gm < M;
      const float* Ap = A + (long)imin(gm, M - 1) * lda;
      #pragma unroll
      for (int u = 0; u < 8; ++u) {
        int gk0 = k0 + aKK + u, gk1 = gk0 + 8;
        float m0f = (rowOk && gk0 < K) ? 1.f : 0.f;
        float m1f = (rowOk && gk1 < K) ? 1.f : 0.f;
        x0[u] = (_Float16)(Ap[imin(gk0, K - 1)] * m0f);
        x1[u] = (_Float16)(Ap[imin(gk1, K - 1)] * m1f);
      }
    }
  };
  auto loadB = [&](int k0, v8h& bv) {
    if (transB) {
      int gn = n0 + bCol;
      if (nFull && k0 + 32 <= K) {
        const float* Bp = B + (long)gn * ldb + (k0 + bKK);
        #pragma unroll
        for (int u = 0; u < 8; ++u) bv[u] = (_Float16)Bp[u];
      } else {
        bool nOk = gn < N;
        const float* Bp = B + (long)imin(gn, N - 1) * ldb;
        #pragma unroll
        for (int u = 0; u < 8; ++u) {
          int gk = k0 + bKK + u;
          float msk = (nOk && gk < K) ? 1.f : 0.f;
          bv[u] = (_Float16)(Bp[imin(gk, K - 1)] * msk);
        }
      }
    } else {
      int gk = k0 + bKrow;
      if (nFull && k0 + 32 <= K) {
        const float* Bp = B + (long)gk * ldb + (n0 + bNN);
        #pragma unroll
        for (int u = 0; u < 8; ++u) bv[u] = (_Float16)Bp[u];
      } else {
        bool kOk = gk < K;
        const float* Bp = B + (long)imin(gk, K - 1) * ldb;
        #pragma unroll
        for (int u = 0; u < 8; ++u) {
          int gn = n0 + bNN + u;
          float msk = (kOk && gn < N) ? 1.f : 0.f;
          bv[u] = (_Float16)(Bp[imin(gn, N - 1)] * msk);
        }
      }
    }
  };

  v8f acc0 = {}, acc1 = {}, acc2 = {}, acc3 = {};

  v8h aR0, aR1, bR;
  loadA(0, aR0, aR1);
  loadB(0, bR);

  for (int k0 = 0; k0 < K; k0 += 32) {
    // ---- commit prefetched registers to LDS ----
    *(v8h*)&As[aRow][aKK]     = aR0;
    *(v8h*)&As[aRow][aKK + 8] = aR1;
    if (transB) {
      *(v8h*)&BsT[bCol][bKK] = bR;
    } else {
      #pragma unroll
      for (int u = 0; u < 8; ++u) BsT[bNN + u][bKrow] = bR[u];
    }
    __syncthreads();

    // ---- prefetch next K-tile into registers (overlaps WMMA below) ----
    int k1 = k0 + 32;
    if (k1 < K) { loadA(k1, aR0, aR1); loadB(k1, bR); }

    // ---- fragments (aligned ds_load_b128) + 4 WMMA sharing A ----
    // A 16x32: lanes 0-15 row=lane K=0..15; lanes 16-31 same rows K=16..31
    // B 32x16: lanes 0-15 col=lane K=0..15; lanes 16-31 K=16..31
    const int r  = lane & 15;
    const int kh = (lane >> 4) * 16;

    v8h alo = *(const v8h*)&As[wave * 16 + r][kh];
    v8h ahi = *(const v8h*)&As[wave * 16 + r][kh + 8];
    v16h af;
    #pragma unroll
    for (int u = 0; u < 8; ++u) { af[u] = alo[u]; af[u + 8] = ahi[u]; }

    #pragma unroll
    for (int t = 0; t < 4; ++t) {
      v8h blo = *(const v8h*)&BsT[t * 16 + r][kh];
      v8h bhi = *(const v8h*)&BsT[t * 16 + r][kh + 8];
      v16h bf;
      #pragma unroll
      for (int u = 0; u < 8; ++u) { bf[u] = blo[u]; bf[u + 8] = bhi[u]; }
      v8f src = (t == 0) ? acc0 : (t == 1) ? acc1 : (t == 2) ? acc2 : acc3;
      v8f dst = __builtin_amdgcn_wmma_f32_16x16x32_f16(false, af, false, bf,
                                                       (short)0, src,
                                                       false, false);
      if (t == 0) acc0 = dst; else if (t == 1) acc1 = dst;
      else if (t == 2) acc2 = dst; else acc3 = dst;
    }
    __syncthreads();
  }

  // --- epilogue: C layout lanes 0-15 -> M=vgpr, lanes 16-31 -> M=8+vgpr ---
  const int r       = lane & 15;
  const int rowBase = m0 + wave * 16 + (lane >> 4) * 8;
  if (mFull && nFull) {
    #pragma unroll
    for (int rr = 0; rr < 8; ++rr) {
      long rowOff = (long)(rowBase + rr) * ldc;
      #pragma unroll
      for (int t = 0; t < 4; ++t) {
        int gn = n0 + t * 16 + r;
        float v = ((t == 0) ? acc0[rr] : (t == 1) ? acc1[rr]
                 : (t == 2) ? acc2[rr] : acc3[rr]) * alpha;
        if (bias) v += bias[gn];
        C[rowOff + gn] = actf(v, act);
      }
    }
  } else {
    #pragma unroll
    for (int rr = 0; rr < 8; ++rr) {
      int gm = rowBase + rr;
      if (gm >= M) continue;
      #pragma unroll
      for (int t = 0; t < 4; ++t) {
        int gn = n0 + t * 16 + r;
        if (gn >= N) continue;
        float v = ((t == 0) ? acc0[rr] : (t == 1) ? acc1[rr]
                 : (t == 2) ? acc2[rr] : acc3[rr]) * alpha;
        if (bias) v += bias[gn];
        C[(long)gm * ldc + gn] = actf(v, act);
      }
    }
  }
}

// ---------------------------------------------------------------------------
// Direct conv2d, stride 2, VALID; one thread per output element.
// ---------------------------------------------------------------------------
__global__ void __launch_bounds__(WGSZ)
conv2d_act(const float* __restrict__ x, const float* __restrict__ w,
           const float* __restrict__ bias, float* __restrict__ y,
           int Nimg, int Cin, int Hin, int Win,
           int Cout, int Hout, int Wout, int kk, int act)
{
  int i = blockIdx.x * blockDim.x + threadIdx.x;
  int total = Nimg * Cout * Hout * Wout;
  if (i >= total) return;
  int ow = i % Wout; int t = i / Wout;
  int oh = t % Hout; t /= Hout;
  int co = t % Cout; int n = t / Cout;

  float acc = bias[co];
  const float* xb = x + (long)n * Cin * Hin * Win;
  const float* wb = w + (long)co * Cin * kk * kk;
  for (int ci = 0; ci < Cin; ++ci) {
    const float* xc = xb + (long)ci * Hin * Win;
    const float* wc = wb + ci * kk * kk;
    for (int kh = 0; kh < kk; ++kh) {
      const float* xr = xc + (long)(oh * 2 + kh) * Win + ow * 2;
      const float* wr = wc + kh * kk;
      for (int kw = 0; kw < kk; ++kw) acc += xr[kw] * wr[kw];
    }
  }
  y[i] = actf(acc, act);
}

// ---------------------------------------------------------------------------
// Transposed conv (lax.conv_transpose VALID, stride 2, OIHW, no kernel flip):
// out[oh] += x[(oh-(k-1)+kh)/2] * w[kh]  when index is integral & in range.
// ---------------------------------------------------------------------------
__global__ void __launch_bounds__(WGSZ)
deconv2d_act(const float* __restrict__ x, const float* __restrict__ w,
             const float* __restrict__ bias, float* __restrict__ y,
             int Nimg, int Cin, int Hin, int Win,
             int Cout, int Hout, int Wout, int kk, int act)
{
  int i = blockIdx.x * blockDim.x + threadIdx.x;
  int total = Nimg * Cout * Hout * Wout;
  if (i >= total) return;
  int ow = i % Wout; int t = i / Wout;
  int oh = t % Hout; t /= Hout;
  int co = t % Cout; int n = t / Cout;

  float acc = bias[co];
  const float* xb = x + (long)n * Cin * Hin * Win;
  const float* wb = w + (long)co * Cin * kk * kk;
  for (int kh = 0; kh < kk; ++kh) {
    int ph = oh - (kk - 1) + kh;
    if (ph < 0 || (ph & 1) || (ph >> 1) >= Hin) continue;
    int ih = ph >> 1;
    for (int kw = 0; kw < kk; ++kw) {
      int pw = ow - (kk - 1) + kw;
      if (pw < 0 || (pw & 1) || (pw >> 1) >= Win) continue;
      int iw = pw >> 1;
      const float* xp = xb + (long)ih * Win + iw;
      const float* wp = wb + kh * kk + kw;
      for (int ci = 0; ci < Cin; ++ci)
        acc += xp[(long)ci * Hin * Win] * wp[(long)ci * kk * kk];
    }
  }
  y[i] = actf(acc, act);
}

// ---------------------------------------------------------------------------
// LayerNorm over last dim 256; one wave32 per row, 8 elems/lane.
// ---------------------------------------------------------------------------
__global__ void __launch_bounds__(WGSZ)
layernorm256(const float* __restrict__ x, const float* __restrict__ g,
             const float* __restrict__ b, float* __restrict__ y, int rows)
{
  int row  = blockIdx.x * 8 + (threadIdx.x >> 5);
  int lane = threadIdx.x & 31;
  if (row >= rows) return;
  const float* xr = x + (long)row * 256;

  float v[8]; float s = 0.f;
  #pragma unroll
  for (int u = 0; u < 8; ++u) { v[u] = xr[lane + u * 32]; s += v[u]; }
  #pragma unroll
  for (int o = 16; o > 0; o >>= 1) s += __shfl_xor(s, o, 32);
  float mean = s * (1.f / 256.f);

  float var = 0.f;
  #pragma unroll
  for (int u = 0; u < 8; ++u) { float d = v[u] - mean; var += d * d; }
  #pragma unroll
  for (int o = 16; o > 0; o >>= 1) var += __shfl_xor(var, o, 32);
  float inv = rsqrtf(var * (1.f / 256.f) + 1e-5f);

  float* yr = y + (long)row * 256;
  #pragma unroll
  for (int u = 0; u < 8; ++u) {
    int c = lane + u * 32;
    yr[c] = (v[u] - mean) * inv * g[c] + b[c];
  }
}

// ---------------------------------------------------------------------------
// Causal softmax over scores[bh][q][0..T-1]; block = one (bh, q) row, T=512.
// ---------------------------------------------------------------------------
__global__ void __launch_bounds__(WGSZ)
softmax_causal(float* __restrict__ s, int T)
{
  int q  = blockIdx.x;
  long base = ((long)blockIdx.y * T + q) * T;
  int tid = threadIdx.x, lane = tid & 31, wv = tid >> 5;
  __shared__ float red[8];

  float vals[2]; float mx = -3.4e38f;
  #pragma unroll
  for (int i = 0; i < 2; ++i) {
    int col = tid + i * 256;
    float v = s[base + col];
    if (col > q) v = -1e10f;
    vals[i] = v; mx = fmaxf(mx, v);
  }
  #pragma unroll
  for (int o = 16; o > 0; o >>= 1) mx = fmaxf(mx, __shfl_xor(mx, o, 32));
  if (lane == 0) red[wv] = mx;
  __syncthreads();
  float bm = red[0];
  #pragma unroll
  for (int i = 1; i < 8; ++i) bm = fmaxf(bm, red[i]);
  __syncthreads();

  float sum = 0.f;
  #pragma unroll
  for (int i = 0; i < 2; ++i) { vals[i] = expf(vals[i] - bm); sum += vals[i]; }
  #pragma unroll
  for (int o = 16; o > 0; o >>= 1) sum += __shfl_xor(sum, o, 32);
  if (lane == 0) red[wv] = sum;
  __syncthreads();
  float bs = 0.f;
  #pragma unroll
  for (int i = 0; i < 8; ++i) bs += red[i];
  float rinv = 1.f / bs;
  #pragma unroll
  for (int i = 0; i < 2; ++i) s[base + tid + i * 256] = vals[i] * rinv;
}

// --------------------------- elementwise helpers ---------------------------
__global__ void reparam_k(const float* mu, const float* lv, const float* eps,
                          float* z, int n) {
  int i = blockIdx.x * blockDim.x + threadIdx.x;
  if (i < n) z[i] = mu[i] + expf(0.5f * lv[i]) * eps[i];
}
__global__ void add_embed_k(const float* z, const float* pos, const float* ae,
                            const float* se, float* x, int n) {
  int i = blockIdx.x * blockDim.x + threadIdx.x;
  if (i >= n) return;
  int col = i & 255, row = i >> 8, t = row & 511;
  x[i] = z[i] + pos[t * 256 + col] + ae[i] + se[i];
}
__global__ void add_inplace_k(float* x, const float* y, int n) {
  int i = blockIdx.x * blockDim.x + threadIdx.x;
  if (i < n) x[i] += y[i];
}

// ===========================================================================
extern "C" void kernel_launch(void* const* d_in, const int* in_sizes, int n_in,
                              void* d_out, int out_size, void* d_ws, size_t ws_size,
                              hipStream_t stream)
{
  (void)in_sizes; (void)n_in; (void)out_size; (void)ws_size;
  auto F = [&](int i) { return (const float*)d_in[i]; };

  const float* img    = F(0);   // (4096, 1, 64, 64)
  const float* state  = F(1);   // (4096, 2)
  const float* action = F(2);   // (4096, 3)
  /* F(3) = dones, unused */
  const float* eps    = F(4);   // (4096, 256)

  int p = 5;
  const float *encw[4], *encb[4];
  for (int i = 0; i < 4; ++i) { encw[i] = F(p++); encb[i] = F(p++); }
  const float* fcmu_w = F(p++); const float* fcmu_b = F(p++);
  const float* fclv_w = F(p++); const float* fclv_b = F(p++);
  const float* dfc_w  = F(p++); const float* dfc_b  = F(p++);
  const float *decw[4], *decb[4];
  for (int i = 0; i < 4; ++i) { decw[i] = F(p++); decb[i] = F(p++); }
  const float* pos  = F(p++);
  const float* aw   = F(p++); const float* ab  = F(p++);
  const float* sw   = F(p++); const float* sb  = F(p++);
  const float* lnfg = F(p++); const float* lnfb = F(p++);
  const float* zhw  = F(p++); const float* zhb = F(p++);
  const float* shw  = F(p++); const float* shb = F(p++);
  struct Blk { const float *ln1g,*ln1b,*ln2g,*ln2b,*qw,*qb,*kw,*kb,*vw,*vb,*pw,*pb,*f1w,*f1b,*f2w,*f2b; };
  Blk blk[6];
  for (int i = 0; i < 6; ++i) {
    blk[i].ln1g=F(p++); blk[i].ln1b=F(p++); blk[i].ln2g=F(p++); blk[i].ln2b=F(p++);
    blk[i].qw=F(p++);  blk[i].qb=F(p++);  blk[i].kw=F(p++);  blk[i].kb=F(p++);
    blk[i].vw=F(p++);  blk[i].vb=F(p++);  blk[i].pw=F(p++);  blk[i].pb=F(p++);
    blk[i].f1w=F(p++); blk[i].f1b=F(p++); blk[i].f2w=F(p++); blk[i].f2b=F(p++);
  }

  const long NTOK = 4096;               // B*T
  const int  T = 512, NB = 8, NH = 8;   // HD = 32
  const int  CHN = 256, NCHUNK = 16;    // image chunking
  const long IMGN = 8L * 512 * 64 * 64; // output image elements

  // ----- workspace carve -----
  char* wsb = (char*)d_ws;
  long cur = 0;
  auto alloc = [&](long nflt) {
    float* r = (float*)(wsb + cur);
    cur = (cur + nflt * 4 + 255) & ~255L;
    return r;
  };
  float* hFlat = alloc(NTOK * 1024);
  float* mu    = alloc(NTOK * 256);
  float* lv    = alloc(NTOK * 256);
  float* zbuf  = alloc(NTOK * 256);
  float* xbuf  = alloc(NTOK * 256);
  float* xn    = alloc(NTOK * 256);
  float* qb_   = alloc(NTOK * 256);
  float* kb_   = alloc(NTOK * 256);
  float* vb_   = alloc(NTOK * 256);
  float* attY  = alloc(NTOK * 256);
  float* tmp   = alloc(NTOK * 256);
  float* f1    = alloc(NTOK * 1024);
  float* scores= alloc(64L * T * T);
  float* zpred = alloc(NTOK * 256);
  float* decfc = alloc(NTOK * 1024);
  float* bufA  = alloc((long)CHN * 32 * 31 * 31);  // >= CHN*32*30*30 too
  float* bufB  = alloc((long)CHN * 64 * 14 * 14);  // >= CHN*64*13*13 too

  float* out = (float*)d_out;

  // ----- launch helpers -----
  auto GEMM = [&](const float* A, const float* Bm, const float* bias, float* Cm,
                  int M, int N, int K, int lda, int ldb, int ldc,
                  int transB, int act, float alpha,
                  int nb1 = 1, int nb2 = 1,
                  long sA1 = 0, long sA2 = 0, long sB1 = 0, long sB2 = 0,
                  long sC1 = 0, long sC2 = 0) {
    dim3 g((unsigned)((N + 63) / 64), (unsigned)((M + 127) / 128),
           (unsigned)(nb1 * nb2));
    gemm_f16wmma<<<g, WGSZ, 0, stream>>>(A, Bm, bias, Cm, M, N, K, lda, ldb, ldc,
                                         sA1, sA2, sB1, sB2, sC1, sC2,
                                         nb2 < 1 ? 1 : nb2, transB, act, alpha);
  };
  auto EWG = [](long n) { return dim3((unsigned)((n + WGSZ - 1) / WGSZ)); };

  // ================= VAE encoder (chunked) =================
  for (int c = 0; c < NCHUNK; ++c) {
    const float* xin = img + (long)c * CHN * 64 * 64;
    long t1 = (long)CHN * 32 * 31 * 31;
    conv2d_act<<<EWG(t1), WGSZ, 0, stream>>>(xin, encw[0], encb[0], bufA,
                                             CHN, 1, 64, 64, 32, 31, 31, 4, 1);
    long t2 = (long)CHN * 64 * 14 * 14;
    conv2d_act<<<EWG(t2), WGSZ, 0, stream>>>(bufA, encw[1], encb[1], bufB,
                                             CHN, 32, 31, 31, 64, 14, 14, 4, 1);
    long t3 = (long)CHN * 128 * 6 * 6;
    conv2d_act<<<EWG(t3), WGSZ, 0, stream>>>(bufB, encw[2], encb[2], bufA,
                                             CHN, 64, 14, 14, 128, 6, 6, 4, 1);
    long t4 = (long)CHN * 256 * 2 * 2;
    conv2d_act<<<EWG(t4), WGSZ, 0, stream>>>(bufA, encw[3], encb[3],
                                             hFlat + (long)c * CHN * 1024,
                                             CHN, 128, 6, 6, 256, 2, 2, 4, 1);
  }
  GEMM(hFlat, fcmu_w, fcmu_b, mu, 4096, 256, 1024, 1024, 256, 256, 0, 0, 1.f);
  GEMM(hFlat, fclv_w, fclv_b, lv, 4096, 256, 1024, 1024, 256, 256, 0, 0, 1.f);
  reparam_k<<<EWG(NTOK * 256), WGSZ, 0, stream>>>(mu, lv, eps, zbuf,
                                                  (int)(NTOK * 256));

  // ================= token embedding =================
  GEMM(action, aw, ab, qb_, 4096, 256, 3, 3, 256, 256, 0, 0, 1.f);
  GEMM(state,  sw, sb, kb_, 4096, 256, 2, 2, 256, 256, 0, 0, 1.f);
  add_embed_k<<<EWG(NTOK * 256), WGSZ, 0, stream>>>(zbuf, pos, qb_, kb_, xbuf,
                                                    (int)(NTOK * 256));

  // ================= transformer =================
  const float scale = 0.17677669529663687f;  // 1/sqrt(32)
  for (int b = 0; b < 6; ++b) {
    const Blk& L = blk[b];
    layernorm256<<<dim3(512), WGSZ, 0, stream>>>(xbuf, L.ln1g, L.ln1b, xn, 4096);
    GEMM(xn, L.qw, L.qb, qb_, 4096, 256, 256, 256, 256, 256, 0, 0, 1.f);
    GEMM(xn, L.kw, L.kb, kb_, 4096, 256, 256, 256, 256, 256, 0, 0, 1.f);
    GEMM(xn, L.vw, L.vb, vb_, 4096, 256, 256, 256, 256, 256, 0, 0, 1.f);

    // scores[b,h] = (Q_bh @ K_bh^T) * scale   (64 batched 512x512x32 GEMMs)
    GEMM(qb_, kb_, nullptr, scores, T, T, 32, 256, 256, T, 1, 0, scale,
         NB, NH, (long)T * 256, 32, (long)T * 256, 32,
         (long)NH * T * T, (long)T * T);
    softmax_causal<<<dim3(T, NB * NH), WGSZ, 0, stream>>>(scores, T);
    // attY[b,h] = P_bh @ V_bh   (64 batched 512x32x512 GEMMs)
    GEMM(scores, vb_, nullptr, attY, T, 32, T, T, 256, 256, 0, 0, 1.f,
         NB, NH, (long)NH * T * T, (long)T * T, (long)T * 256, 32,
         (long)T * 256, 32);

    GEMM(attY, L.pw, L.pb, tmp, 4096, 256, 256, 256, 256, 256, 0, 0, 1.f);
    add_inplace_k<<<EWG(NTOK * 256), WGSZ, 0, stream>>>(xbuf, tmp,
                                                        (int)(NTOK * 256));
    layernorm256<<<dim3(512), WGSZ, 0, stream>>>(xbuf, L.ln2g, L.ln2b, xn, 4096);
    GEMM(xn, L.f1w, L.f1b, f1, 4096, 1024, 256, 256, 1024, 1024, 0, 2, 1.f);
    GEMM(f1, L.f2w, L.f2b, tmp, 4096, 256, 1024, 1024, 256, 256, 0, 0, 1.f);
    add_inplace_k<<<EWG(NTOK * 256), WGSZ, 0, stream>>>(xbuf, tmp,
                                                        (int)(NTOK * 256));
  }

  // ================= heads =================
  layernorm256<<<dim3(512), WGSZ, 0, stream>>>(xbuf, lnfg, lnfb, xn, 4096);
  GEMM(xn, zhw, zhb, zpred, 4096, 256, 256, 256, 256, 256, 0, 0, 1.f);
  GEMM(xn, shw, shb, out + IMGN, 4096, 2, 256, 256, 2, 2, 0, 0, 1.f);

  // ================= VAE decoder (chunked) =================
  GEMM(zpred, dfc_w, dfc_b, decfc, 4096, 1024, 256, 256, 1024, 1024, 0, 0, 1.f);
  for (int c = 0; c < NCHUNK; ++c) {
    const float* hin = decfc + (long)c * CHN * 1024;  // (CHN, 1024, 1, 1)
    long t1 = (long)CHN * 128 * 5 * 5;
    deconv2d_act<<<EWG(t1), WGSZ, 0, stream>>>(hin, decw[0], decb[0], bufA,
                                               CHN, 1024, 1, 1, 128, 5, 5, 5, 1);
    long t2 = (long)CHN * 64 * 13 * 13;
    deconv2d_act<<<EWG(t2), WGSZ, 0, stream>>>(bufA, decw[1], decb[1], bufB,
                                               CHN, 128, 5, 5, 64, 13, 13, 5, 1);
    long t3 = (long)CHN * 32 * 30 * 30;
    deconv2d_act<<<EWG(t3), WGSZ, 0, stream>>>(bufB, decw[2], decb[2], bufA,
                                               CHN, 64, 13, 13, 32, 30, 30, 6, 1);
    long t4 = (long)CHN * 1 * 64 * 64;
    deconv2d_act<<<EWG(t4), WGSZ, 0, stream>>>(bufA, decw[3], decb[3],
                                               out + (long)c * CHN * 64 * 64,
                                               CHN, 32, 30, 30, 1, 64, 64, 6, 3);
  }
}
